// KNNSpatialEncoder_5540507812268
// MI455X (gfx1250) — matrix operand order repaired and checked by reading, not compiled, not run on hardware
//
#include <hip/hip_runtime.h>
#include <hip/hip_bf16.h>
#include <math.h>

#define K_TOTAL 48
#define RING_K 16
#define NUM_HEADS 4
#define RADIUS 50.0f
#define Nn 4096
#define Bb 2

typedef __attribute__((ext_vector_type(16))) __bf16 v16bf;
typedef __attribute__((ext_vector_type(8)))  __bf16 v8bf;
typedef __attribute__((ext_vector_type(8)))  float  v8f;

// ---------- WMMA fragment helpers (wave32 layouts per CDNA5 ISA 7.12.2) ----------
// A 16x32 bf16: lane L row M=L%16; halves 0..7 -> K=kBase+s+i, halves 8..15 -> K=kBase+16+s+i,
// s = (L<16 ? 0 : 8). Two aligned 16B loads.
__device__ inline v16bf frag_a_bf16(const __bf16* A, int lda, int mBase, int kBase, int lane) {
  int m = mBase + (lane & 15);
  int s = (lane < 16) ? 0 : 8;
  const __bf16* p = A + (long)m * lda + kBase + s;
  v8bf lo = *(const v8bf*)p;
  v8bf hi = *(const v8bf*)(p + 16);
  v16bf a;
#pragma unroll
  for (int i = 0; i < 8; ++i) { a[i] = lo[i]; a[i + 8] = hi[i]; }
  return a;
}

// B KxN bf16 with B[k][o] = W[o][k] (x @ W.T): lane L column o=oBase+L%16,
// halves 0..7 -> K = kBase+s+i, halves 8..15 -> K = kBase+16+s+i, s = (L<16?0:8) => two aligned 16B loads.
__device__ inline v16bf frag_b_bf16(const __bf16* W, int ldw, int oBase, int kBase, int lane) {
  int o  = oBase + (lane & 15);
  int k0 = kBase + ((lane < 16) ? 0 : 8);
  const __bf16* p = W + (long)o * ldw + k0;
  v8bf lo = *(const v8bf*)p;
  v8bf hi = *(const v8bf*)(p + 16);
  v16bf b;
#pragma unroll
  for (int i = 0; i < 8; ++i) { b[i] = lo[i]; b[i + 8] = hi[i]; }
  return b;
}

// ---------- prep: fp32 -> bf16 bulk convert (4 elems/thread) ----------
__global__ void f32_to_bf16_kernel(const float* __restrict__ src, __bf16* __restrict__ dst, int n) {
  int i = (blockIdx.x * blockDim.x + threadIdx.x) * 4;
  if (i < n) {
    float4 f = *(const float4*)(src + i);
    dst[i + 0] = (__bf16)f.x; dst[i + 1] = (__bf16)f.y;
    dst[i + 2] = (__bf16)f.z; dst[i + 3] = (__bf16)f.w;
  }
}

// ---------- generic [M x 256] @ W[256x256].T + bias (all bf16, pipelined B) ----------
template<bool WF, bool WB>
__global__ void gemm256_wmma(const __bf16* __restrict__ A, const __bf16* __restrict__ W,
                             const float* __restrict__ bias, float* __restrict__ Cf,
                             __bf16* __restrict__ Cb, int M, float scale) {
  int lane  = threadIdx.x & 31;
  int gwave = blockIdx.x * (blockDim.x >> 5) + (threadIdx.x >> 5);
  int mBase = gwave * 16;
  if (mBase >= M) return;  // wave-uniform
  v16bf afrag[8];
#pragma unroll
  for (int ks = 0; ks < 8; ++ks) afrag[ks] = frag_a_bf16(A, 256, mBase, ks * 32, lane);
  v16bf bcur = frag_b_bf16(W, 256, 0, 0, lane);   // one-ahead pipeline
  for (int nBase = 0; nBase < 256; nBase += 16) {
    float bv = bias[nBase + (lane & 15)];
    v8f acc;
#pragma unroll
    for (int r = 0; r < 8; ++r) acc[r] = bv;
#pragma unroll
    for (int ks = 0; ks < 8; ++ks) {
      int tn = (ks == 7) ? ((nBase + 16) & 255) : nBase;   // wrap: 1 redundant load at end
      int tk = ((ks + 1) & 7) * 32;
      v16bf bnxt = frag_b_bf16(W, 256, tn, tk, lane);
      acc = __builtin_amdgcn_wmma_f32_16x16x32_bf16(false, afrag[ks], false, bcur, (short)0, acc, false, false);
      bcur = bnxt;
    }
    int col    = nBase + (lane & 15);
    int rowOff = mBase + ((lane < 16) ? 0 : 8);
#pragma unroll
    for (int r = 0; r < 8; ++r) {
      float v = acc[r] * scale;
      if (WF) Cf[(long)(rowOff + r) * 256 + col] = v;
      if (WB) Cb[(long)(rowOff + r) * 256 + col] = (__bf16)v;
    }
  }
}

// ---------- KNN: per-thread stable top-48 over causally/radius-masked distances ----------
__global__ void knn_kernel(const float* __restrict__ coords, int* __restrict__ idxOut,
                           float* __restrict__ kdOut) {
  const int TILE = 512;
  __shared__ float sc[TILE * 3];
  int gid = blockIdx.x * blockDim.x + threadIdx.x;
  int b = gid / Nn;
  int i = gid - b * Nn;
  const float* cb = coords + (long)b * Nn * 3;
  float xi = cb[i * 3 + 0], yi = cb[i * 3 + 1], ti = cb[i * 3 + 2];
  float kd[K_TOTAL]; int ki[K_TOTAL]; int cnt = 0;
  for (int t0 = 0; t0 < Nn; t0 += TILE) {
    __syncthreads();
    for (int s = threadIdx.x; s < TILE * 3; s += blockDim.x) sc[s] = cb[t0 * 3 + s];
    __syncthreads();
    for (int jj = 0; jj < TILE; ++jj) {
      int j = t0 + jj;
      float tj = sc[jj * 3 + 2];
      float dx = xi - sc[jj * 3 + 0];
      float dy = yi - sc[jj * 3 + 1];
      float sd = sqrtf(dx * dx + dy * dy);
      if (j == i || tj > ti || sd > RADIUS) continue;
      float d = sd + 0.3f * fabsf(ti - tj);
      if (cnt == K_TOTAL && d >= kd[K_TOTAL - 1]) continue;   // strict <: keeps top_k tie order
      int p = (cnt < K_TOTAL) ? cnt++ : (K_TOTAL - 1);
      while (p > 0 && kd[p - 1] > d) { kd[p] = kd[p - 1]; ki[p] = ki[p - 1]; --p; }
      kd[p] = d; ki[p] = j;
    }
  }
  long base = (long)gid * K_TOTAL;
  for (int s = 0; s < K_TOTAL; ++s) {
    kdOut[base + s]  = (s < cnt) ? kd[s] : __builtin_inff();
    idxOut[base + s] = (s < cnt) ? ki[s] : 0;
  }
}

// ---------- fused per-point: async row gather + PE (WMMA) + k/v WMMA + attention ----------
__global__ void encode_node_kernel(const float* __restrict__ coords,
                                   const __bf16* __restrict__ featprojb,
                                   const float* __restrict__ qbuf,
                                   const int* __restrict__ idxBuf,
                                   const float* __restrict__ kdBuf,
                                   const float* __restrict__ Wpe1, const float* __restrict__ bpe1,
                                   const __bf16* __restrict__ Wpe2b, const float* __restrict__ bpe2,
                                   const __bf16* __restrict__ Winb, const float* __restrict__ bin_,
                                   __bf16* __restrict__ attnOutB) {
  extern __shared__ char smem[];
  __bf16* rowbf = (__bf16*)smem;                                 // 48*256*2   = 24576 B
  float*  kvbuf = (float*)(smem + 48 * 256 * 2);                 // 2*48*256*4 = 98304 B (k then v)
  __bf16* h_bf  = (__bf16*)kvbuf;                                // alias: dead before k/v written
  int*    sIdx  = (int*)(smem + 48 * 256 * 2 + 2 * 48 * 256 * 4);
  float*  sKd   = (float*)((char*)sIdx + 48 * 4);
  float*  sVal  = (float*)((char*)sKd + 48 * 4);
  float*  sRel  = (float*)((char*)sVal + 48 * 4);                // 48*4 floats

  int bn = blockIdx.x;
  int b = bn / Nn;
  int tid = threadIdx.x, lane = tid & 31, wave = tid >> 5;
  const float* cb = coords + (long)b * Nn * 3;
  int n = bn - b * Nn;
  float xi = cb[n * 3], yi = cb[n * 3 + 1], ti = cb[n * 3 + 2];
  long kb = (long)bn * K_TOTAL;

  if (tid < K_TOTAL) {
    int   j  = idxBuf[kb + tid];
    float kr = kdBuf[kb + tid];
    float v  = (kr < 1e30f) ? 1.f : 0.f;
    float kc = (v > 0.f) ? kr : 0.f;
    sIdx[tid] = j; sVal[tid] = v; sKd[tid] = kc;
    sRel[tid * 4 + 0] = cb[j * 3 + 0] - xi;
    sRel[tid * 4 + 1] = cb[j * 3 + 1] - yi;
    sRel[tid * 4 + 2] = cb[j * 3 + 2] - ti;
    sRel[tid * 4 + 3] = kc;
  }
  __syncthreads();

  // Async DMA of the 48 gathered feat_proj rows (bf16) into rowbf (1536 x 16B chunks),
  // overlapped with the PE hidden-layer compute below.
  unsigned ldsBase = (unsigned)(size_t)rowbf;     // low 32 bits of generic LDS ptr = LDS offset
  for (int t = tid; t < 48 * 32; t += blockDim.x) {
    int r = t >> 5, c = t & 31;                   // row r, 16B chunk c (32 chunks of 8 bf16)
    const __bf16* gp = featprojb + ((long)b * Nn + sIdx[r]) * 256 + c * 8;
    unsigned lo = ldsBase + (unsigned)t * 16;
    asm volatile("global_load_async_to_lds_b128 %0, %1, off" :: "v"(lo), "v"(gp) : "memory");
  }

  // PE hidden layer: 48 rows x 128 hidden -> bf16 LDS (h_bf region; no overlap with rowbf DMA)
  for (int t = tid; t < 48 * 128; t += blockDim.x) {
    int r = t >> 7, j = t & 127;
    float h = bpe1[j];
#pragma unroll
    for (int c = 0; c < 4; ++c) h += sRel[r * 4 + c] * Wpe1[j * 4 + c];
    h_bf[t] = (__bf16)((h > 0.f) ? h : 0.f);
  }
  asm volatile("s_wait_asynccnt 0x0" ::: "memory");   // my async rows are in LDS
  __syncthreads();                                    // everyone's rows + hidden visible

  // PE output layer on the matrix pipe: pe = h[48x128] @ Wpe2.T + bpe2, added onto
  // the DMA-gathered rows. 48 tile-jobs = rowTile(3) x colTile(16), K=128 -> 4 k-steps.
  for (int job = wave; job < 48; job += (blockDim.x >> 5)) {
    int rt = job / 16, ct = job - rt * 16;
    int nBase = ct * 16;
    float bv = bpe2[nBase + (lane & 15)];
    v8f acc;
#pragma unroll
    for (int r = 0; r < 8; ++r) acc[r] = bv;
#pragma unroll
    for (int ks = 0; ks < 4; ++ks) {
      v16bf a  = frag_a_bf16(h_bf, 128, rt * 16, ks * 32, lane);     // LDS source
      v16bf bf = frag_b_bf16(Wpe2b, 128, nBase, ks * 32, lane);
      acc = __builtin_amdgcn_wmma_f32_16x16x32_bf16(false, a, false, bf, (short)0, acc, false, false);
    }
    int col    = nBase + (lane & 15);
    int rowOff = rt * 16 + ((lane < 16) ? 0 : 8);
#pragma unroll
    for (int r = 0; r < 8; ++r) {
      int idx = (rowOff + r) * 256 + col;
      rowbf[idx] = (__bf16)((float)rowbf[idx] + acc[r]);
    }
  }
  __syncthreads();   // rowbf final; h_bf region may now be overwritten by k/v

  // k/v projection: 96 WMMA tile-jobs = proj(2) x rowTile(3) x colTile(16), 8 k-steps,
  // with a one-ahead pipelined B fragment (wrap -> 1 redundant load per job).
  for (int job = wave; job < 96; job += (blockDim.x >> 5)) {
    int proj = job / 48;
    int rem  = job - proj * 48;
    int rt   = rem / 16, ct = rem - rt * 16;
    const __bf16* Wp = Winb + (long)(256 + proj * 256) * 256;   // Wk rows 256..511, Wv rows 512..767
    const float*  bp = bin_ + 256 + proj * 256;
    int nBase = ct * 16;
    float bv = bp[nBase + (lane & 15)];
    v8f acc;
#pragma unroll
    for (int r = 0; r < 8; ++r) acc[r] = bv;
    v16bf bcur = frag_b_bf16(Wp, 256, nBase, 0, lane);
#pragma unroll
    for (int ks = 0; ks < 8; ++ks) {
      v16bf bnxt = frag_b_bf16(Wp, 256, nBase, ((ks + 1) & 7) * 32, lane);
      v16bf a    = frag_a_bf16(rowbf, 256, rt * 16, ks * 32, lane);  // LDS source
      acc = __builtin_amdgcn_wmma_f32_16x16x32_bf16(false, a, false, bcur, (short)0, acc, false, false);
      bcur = bnxt;
    }
    int col    = nBase + (lane & 15);
    int rowOff = rt * 16 + ((lane < 16) ? 0 : 8);
    float* dst = kvbuf + proj * 48 * 256;
#pragma unroll
    for (int r = 0; r < 8; ++r) dst[(rowOff + r) * 256 + col] = acc[r];
  }
  __syncthreads();

  // attention: 12 (ring,head) tasks, one wave each; 2 dims per lane of hd=64
  const float* qn = qbuf + (long)bn * 256;   // q pre-scaled by 1/sqrt(hd)
  for (int task = wave; task < 12; task += (blockDim.x >> 5)) {
    int rr = task / NUM_HEADS, h = task - rr * NUM_HEADS;
    int d0 = lane * 2;
    float q0 = qn[h * 64 + d0], q1 = qn[h * 64 + d0 + 1];
    float logits[RING_K];
#pragma unroll
    for (int kk = 0; kk < RING_K; ++kk) {
      int row = rr * RING_K + kk;
      const float* krow = kvbuf + row * 256 + h * 64;
      float part = q0 * krow[d0] + q1 * krow[d0 + 1];
#pragma unroll
      for (int off = 16; off > 0; off >>= 1) part += __shfl_xor(part, off, 32);
      logits[kk] = (sVal[row] > 0.5f) ? part : -1e9f;
    }
    float m = logits[0];
#pragma unroll
    for (int kk = 1; kk < RING_K; ++kk) m = fmaxf(m, logits[kk]);
    float ssum = 0.f;
#pragma unroll
    for (int kk = 0; kk < RING_K; ++kk) { logits[kk] = __expf(logits[kk] - m); ssum += logits[kk]; }
    float inv = 1.f / ssum;
    float o0 = 0.f, o1 = 0.f;
#pragma unroll
    for (int kk = 0; kk < RING_K; ++kk) {
      int row = rr * RING_K + kk;
      const float* vrow = kvbuf + 48 * 256 + row * 256 + h * 64;
      float a = logits[kk] * inv;
      o0 += a * vrow[d0]; o1 += a * vrow[d0 + 1];
    }
    long ob = ((long)bn * 3 + rr) * 256 + h * 64 + d0;
    attnOutB[ob] = (__bf16)o0; attnOutB[ob + 1] = (__bf16)o1;   // bf16: feeds Wout WMMA directly
  }
}

// ---------- gating + fuse + layernorm: one wave per point ----------
__global__ void finalize_kernel(const float* __restrict__ coords,
                                const float* __restrict__ featproj,
                                const float* __restrict__ aggs,
                                const int* __restrict__ idxBuf,
                                const float* __restrict__ kdBuf,
                                const float* __restrict__ Wg1, const float* __restrict__ bg1,
                                const float* __restrict__ Wg2, const float* __restrict__ bg2,
                                const float* __restrict__ gamma, const float* __restrict__ beta,
                                float* __restrict__ out) {
  int bn = blockIdx.x;
  int b = bn / Nn;
  int lane = threadIdx.x;
  const float* cb = coords + (long)b * Nn * 3;
  int n = bn - b * Nn;
  float ti = cb[n * 3 + 2];
  long kb = (long)bn * K_TOTAL;

  float kcl[K_TOTAL], dtv[K_TOTAL]; bool val[K_TOTAL];
  int vc0 = 0; bool any0 = false, any1 = false, any2 = false;
  for (int s = 0; s < K_TOTAL; ++s) {
    float kr = kdBuf[kb + s];
    bool v = kr < 1e30f;
    val[s] = v; kcl[s] = v ? kr : 0.f;
    if (v) { vc0++; if (s < 16) any0 = true; else if (s < 32) any1 = true; else any2 = true; }
    int j = idxBuf[kb + s];
    dtv[s] = fabsf(cb[j * 3 + 2] - ti);
  }
  int vc = (vc0 > 0) ? vc0 : 1;
  float dens = kcl[vc - 1];
  float ratio = fminf(fmaxf((float)vc / 48.f, 0.f), 1.f);
  float dsum = 0.f;
  for (int s = 0; s < K_TOTAL; ++s) if (val[s]) dsum += dtv[s];
  float dmean = dsum / (float)vc;
  float vsum = 0.f;
  for (int s = 0; s < K_TOTAL; ++s) if (val[s]) { float z = dtv[s] - dmean; vsum += z * z; }
  float dstd = sqrtf(vsum / (float)vc + 1e-8f);
  // _safe_norm over a size-1 last axis == clip(x/(x+1e-6), 0, 10)
  auto sn = [](float x) { float y = x / (x + 1e-6f); return fminf(fmaxf(y, 0.f), 10.f); };
  float gi[4] = { sn(dens), ratio, sn(dmean), sn(dstd) };

  float glog[3] = { bg2[0], bg2[1], bg2[2] };
  for (int j = 0; j < 32; ++j) {
    float h = bg1[j];
#pragma unroll
    for (int c = 0; c < 4; ++c) h += gi[c] * Wg1[j * 4 + c];
    h = (h > 0.f) ? h : 0.f;
#pragma unroll
    for (int r = 0; r < 3; ++r) glog[r] += h * Wg2[r * 32 + j];
  }
  bool anyr[3] = { any0, any1, any2 };
  bool nonb = !(any0 || any1 || any2);
#pragma unroll
  for (int r = 0; r < 3; ++r) {
    if (nonb)          glog[r] = (r == 0) ? 10.f : 0.f;
    else if (!anyr[r]) glog[r] = -__builtin_inff();
  }
  float m  = fmaxf(glog[0], fmaxf(glog[1], glog[2]));
  float e0 = __expf(glog[0] - m), e1 = __expf(glog[1] - m), e2 = __expf(glog[2] - m);
  float wsum = e0 + e1 + e2;
  float w0 = e0 / wsum, w1 = e1 / wsum, w2 = e2 / wsum;

  long ab = (long)bn * 3 * 256;
  float x8[8];
  float ssum = 0.f, ssq = 0.f;
#pragma unroll
  for (int i2 = 0; i2 < 8; ++i2) {
    int col = lane + i2 * 32;
    float f = 0.f;
    if (any0) f += w0 * aggs[ab + 0 * 256 + col];
    if (any1) f += w1 * aggs[ab + 1 * 256 + col];
    if (any2) f += w2 * aggs[ab + 2 * 256 + col];
    float x = featproj[(long)bn * 256 + col] + f;
    x8[i2] = x; ssum += x; ssq += x * x;
  }
  for (int off = 16; off > 0; off >>= 1) {
    ssum += __shfl_xor(ssum, off, 32);
    ssq  += __shfl_xor(ssq,  off, 32);
  }
  float mu   = ssum / 256.f;
  float var  = ssq / 256.f - mu * mu;
  float rstd = rsqrtf(var + 1e-5f);
#pragma unroll
  for (int i2 = 0; i2 < 8; ++i2) {
    int col = lane + i2 * 32;
    out[(long)bn * 256 + col] = (x8[i2] - mu) * rstd * gamma[col] + beta[col];
  }
}

extern "C" void kernel_launch(void* const* d_in, const int* in_sizes, int n_in,
                              void* d_out, int out_size, void* d_ws, size_t ws_size,
                              hipStream_t stream) {
  (void)in_sizes; (void)n_in; (void)out_size; (void)ws_size;
  const float* features = (const float*)d_in[0];
  const float* coords   = (const float*)d_in[1];
  const float* Wfeat = (const float*)d_in[2];
  const float* bfeat = (const float*)d_in[3];
  const float* Wpe1  = (const float*)d_in[4];
  const float* bpe1  = (const float*)d_in[5];
  const float* Wpe2  = (const float*)d_in[6];
  const float* bpe2  = (const float*)d_in[7];
  const float* Win   = (const float*)d_in[8];
  const float* bin_  = (const float*)d_in[9];
  const float* Wout  = (const float*)d_in[10];
  const float* bout  = (const float*)d_in[11];
  const float* Wg1   = (const float*)d_in[12];
  const float* bg1   = (const float*)d_in[13];
  const float* Wg2   = (const float*)d_in[14];
  const float* bg2   = (const float*)d_in[15];
  const float* gamma = (const float*)d_in[16];
  const float* beta  = (const float*)d_in[17];

  char* ws = (char*)d_ws;
  float*  featproj  = (float*) (ws);               //  8 MB   [B*N,256] f32
  __bf16* featprojb = (__bf16*)(ws +  8388608);    //  4 MB   [B*N,256] bf16
  float*  qbuf      = (float*) (ws + 12582912);    //  8 MB   [B*N,256] f32 (pre-scaled 1/8)
  int*    idxBuf    = (int*)   (ws + 20971520);    //  1.5 MB [B*N,48]
  float*  kdBuf     = (float*) (ws + 22544384);    //  1.5 MB [B*N,48] (inf = invalid)
  __bf16* attnB     = (__bf16*)(ws + 24117248);    // 12 MB   [B*N,3,256] bf16
  float*  aggsBuf   = (float*) (ws + 36700160);    // 24 MB   [B*N,3,256] f32
  __bf16* featuresb = (__bf16*)(ws + 61865984);    //  4 MB   [B*N,256] bf16
  __bf16* Wfeatb    = (__bf16*)(ws + 66060288);    // 128 KB
  __bf16* Winb      = (__bf16*)(ws + 66191360);    // 384 KB  [768,256]
  __bf16* Woutb     = (__bf16*)(ws + 66584576);    // 128 KB
  __bf16* Wpe2b     = (__bf16*)(ws + 66715648);    //  64 KB  [256,128]

  const int BN = Bb * Nn;           // 8192
  // 0. one-time bf16 conversions of all WMMA operands
  f32_to_bf16_kernel<<<(BN * 256 / 4 + 255) / 256, 256, 0, stream>>>(features, featuresb, BN * 256);
  f32_to_bf16_kernel<<<(256 * 256 / 4 + 255) / 256, 256, 0, stream>>>(Wfeat, Wfeatb, 256 * 256);
  f32_to_bf16_kernel<<<(768 * 256 / 4 + 255) / 256, 256, 0, stream>>>(Win, Winb, 768 * 256);
  f32_to_bf16_kernel<<<(256 * 256 / 4 + 255) / 256, 256, 0, stream>>>(Wout, Woutb, 256 * 256);
  f32_to_bf16_kernel<<<(256 * 128 / 4 + 255) / 256, 256, 0, stream>>>(Wpe2, Wpe2b, 256 * 128);
  // 1. feat_proj = features @ Wfeat.T + bfeat   (f32 + bf16 outputs)
  gemm256_wmma<true, true><<<128, 128, 0, stream>>>(featuresb, Wfeatb, bfeat,
                                                    featproj, featprojb, BN, 1.0f);
  // 2. KNN top-48 (independent)
  knn_kernel<<<BN / 128, 128, 0, stream>>>(coords, idxBuf, kdBuf);
  // 3. q = (feat_proj @ Wq.T + bq) / sqrt(64)   (Wq = Win rows 0..255)
  gemm256_wmma<true, false><<<128, 128, 0, stream>>>(featprojb, Winb, bin_,
                                                     qbuf, (__bf16*)nullptr, BN, 0.125f);
  // 4. fused async gather + PE WMMA + k/v WMMA + attention
  size_t smem = 48 * 256 * 2 + 2 * 48 * 256 * 4 + 3 * (48 * 4) + 48 * 16; // 124224 B
  (void)hipFuncSetAttribute(reinterpret_cast<const void*>(encode_node_kernel),
                            hipFuncAttributeMaxDynamicSharedMemorySize, (int)smem);
  encode_node_kernel<<<BN, 128, smem, stream>>>(coords, featprojb, qbuf, idxBuf, kdBuf,
                                                Wpe1, bpe1, Wpe2b, bpe2, Winb, bin_, attnB);
  // 5. aggs = attn_out @ Wout.T + bout
  gemm256_wmma<true, false><<<384, 128, 0, stream>>>(attnB, Woutb, bout,
                                                     aggsBuf, (__bf16*)nullptr, BN * 3, 1.0f);
  // 6. gating + residual + layernorm
  finalize_kernel<<<BN, 32, 0, stream>>>(coords, featproj, aggsBuf, idxBuf, kdBuf,
                                         Wg1, bg1, Wg2, bg2, gamma, beta, (float*)d_out);
}
